// DifferentiableRenderer_9019431322196
// MI455X (gfx1250) — matrix operand order repaired and checked by reading, not compiled, not run on hardware
//
#include <hip/hip_runtime.h>
#include <hip/hip_bf16.h>

typedef __attribute__((ext_vector_type(16))) _Float16 v16h;
typedef __attribute__((ext_vector_type(8)))  float    v8f;

#define IMG_W   128
#define IMG_HW  16384        // 128*128
#define FX_     150.0f
#define CX_     64.0f
#define EPS_    1e-8f
#define GCHUNK  512          // gaussians staged in LDS per block iteration
#define NSUB    (GCHUNK / 32)

// pack two floats into one dword of two f16 (lowers to v_cvt_pk_f16_f32)
static __device__ inline unsigned pack2h(float a, float b) {
    unsigned short ua = __builtin_bit_cast(unsigned short, (_Float16)a);
    unsigned short ub = __builtin_bit_cast(unsigned short, (_Float16)b);
    return (unsigned)ua | ((unsigned)ub << 16);
}

// A-fragment K mapping (16-bit A 16x32): VGPR j, lane half 'hi'
static __device__ inline int kaIdx(int j, int hi) {
    return ((j & 3) << 1) + ((j >> 2) << 4) + (hi << 3);
}

// ---------------------------------------------------------------------------
// Kernel 1: per-gaussian preprocessing.
// Computes R(q), camera transform, pinhole projection, and the exp2 coefficient.
// Stores float4 {proj_x, proj_y, opacity, -0.5*log2(e)/var} into workspace.
// ---------------------------------------------------------------------------
__global__ __launch_bounds__(256) void splat_prep_kernel(
    const float* __restrict__ pos, const float* __restrict__ opac,
    const float* __restrict__ scal, const float* __restrict__ qvec,
    const float* __restrict__ tvec, float4* __restrict__ g4, int N)
{
    int i = blockIdx.x * blockDim.x + threadIdx.x;
    if (i >= N) return;
    float qw = qvec[0], qx = qvec[1], qy = qvec[2], qz = qvec[3];
    float inv = rsqrtf(qw*qw + qx*qx + qy*qy + qz*qz);
    qw *= inv; qx *= inv; qy *= inv; qz *= inv;
    float r00 = 1.f - 2.f*(qy*qy + qz*qz), r01 = 2.f*(qx*qy - qz*qw), r02 = 2.f*(qx*qz + qy*qw);
    float r10 = 2.f*(qx*qy + qz*qw), r11 = 1.f - 2.f*(qx*qx + qz*qz), r12 = 2.f*(qy*qz - qx*qw);
    float r20 = 2.f*(qx*qz - qy*qw), r21 = 2.f*(qy*qz + qx*qw), r22 = 1.f - 2.f*(qx*qx + qy*qy);
    float px = pos[3*i+0], py = pos[3*i+1], pz = pos[3*i+2];
    float cx = r00*px + r01*py + r02*pz + tvec[0];
    float cy = r10*px + r11*py + r12*pz + tvec[1];
    float cz = r20*px + r21*py + r22*pz + tvec[2];
    float iz = 1.0f / cz;                       // z in [~2,6], safe
    float ux = cx * iz * FX_ + CX_;
    float uy = cy * iz * FX_ + CX_;
    float s  = scal[i];
    float coef = -0.5f * 1.4426950408889634f / (s * s);   // exp(x)=exp2(x*log2e)
    g4[i] = make_float4(ux, uy, opac[i], coef);
}

// ---------------------------------------------------------------------------
// Kernel 2: render. Block = 128 threads = 4 waves. Each wave owns 32
// consecutive pixels (two 16-pixel WMMA tiles sharing one B fragment).
// Per GCHUNK: (a) stage params {px,py,op,coef} + colors {r,g,b,1} to LDS;
// (b) cooperatively pre-swizzle the f16 B fragments (cols r,g,b,1,0..0) into
// LDS in the exact per-lane WMMA register layout, once per block (amortized
// over 4 waves). Hot loop per 32-gaussian group: 1x 32B LDS read for B
// (2x ds_load_b128, hidden under exps), double-buffered ds_load_b128 param
// reads (prefetched one j-iteration ahead), 32x v_exp_f32 -> packed f16 A
// fragments, and two v_wmma_f32_16x16x32_f16 accumulating in f32.
// ---------------------------------------------------------------------------
__global__ __launch_bounds__(128) void splat_render_kernel(
    const float4* __restrict__ g4, const float* __restrict__ colors,
    float* __restrict__ out, int N)
{
    __shared__ float4 sG[GCHUNK];                             // {px, py, op, coef}
    __shared__ float4 sC4[GCHUNK];                            // {r, g, b, 1.0}
    __shared__ __attribute__((aligned(32))) unsigned bfrag[NSUB][32][8];

    const int tid  = threadIdx.x;
    const int lane = tid & 31;
    const int hi   = lane >> 4;        // 0: lanes 0-15, 1: lanes 16-31
    const int col  = lane & 15;        // B/D column (output channel)
    const int waveId  = blockIdx.x * 4 + (tid >> 5);
    const int pixBase = waveId * 32;   // 32 consecutive flat pixels (same row)

    // pixel coordinates for this lane's A-matrix row (M = lane&15) per tile
    const int p0i = pixBase + (lane & 15);
    const float X0 = (float)(p0i & (IMG_W - 1));
    const float X1 = X0 + 16.0f;
    const float Y  = (float)(p0i >> 7);

    const float* sC4f = (const float*)sC4;

    v8f acc0 = {};   // tile0: pixels pixBase..+15  (cols: r,g,b,den,0...)
    v8f acc1 = {};   // tile1: pixels pixBase+16..+31

    for (int gb = 0; gb < N; gb += GCHUNK) {
        __syncthreads();
        for (int i = tid; i < GCHUNK; i += 128) {
            sG[i]  = g4[gb + i];
            const float* cp = colors + (size_t)(gb + i) * 3;
            sC4[i] = make_float4(cp[0], cp[1], cp[2], 1.0f);
        }
        __syncthreads();

        // Pre-swizzle B fragments: bfrag[sub][l][d] = pack(B(k,col_l), B(k+1,col_l))
        // with k = sub*32 + 2(d&3) + 16*(d>=4) + 8*(l>=16), col_l = l&15.
        for (int idx = tid; idx < NSUB * 256; idx += 128) {
            int sub = idx >> 8;
            int l   = (idx >> 3) & 31;
            int d   = idx & 7;
            int k   = sub * 32 + ((d & 3) << 1) + ((d >> 2) << 4) + ((l >> 4) << 3);
            int lc  = l & 15;
            int cc  = (lc < 3) ? lc : 3;                 // clamped, in-bounds
            float s = (lc < 4) ? 1.0f : 0.0f;            // zero cols 4..15
            float v0 = sC4f[(k + 0) * 4 + cc] * s;
            float v1 = sC4f[(k + 1) * 4 + cc] * s;
            bfrag[sub][l][d] = pack2h(v0, v1);
        }
        __syncthreads();

        for (int sub = 0; sub < NSUB; ++sub) {
            const int base = sub * 32;
            // whole B fragment: one 32-byte LDS read, hidden under the exps
            v16h b = *(const v16h*)&bfrag[sub][lane][0];
            v16h a0, a1;
            // software-pipelined param loads: prefetch j+1 while computing j
            float4 gA = sG[base + kaIdx(0, hi) + 0];
            float4 gB = sG[base + kaIdx(0, hi) + 1];
            #pragma unroll
            for (int j = 0; j < 8; ++j) {
                float4 cA = gA, cB = gB;
                if (j < 7) {
                    const int kn = kaIdx(j + 1, hi);
                    gA = sG[base + kn + 0];
                    gB = sG[base + kn + 1];
                }
                {
                    float dy  = cA.y - Y;
                    float dy2 = dy * dy;
                    float dx0 = cA.x - X0;
                    float dx1 = cA.x - X1;
                    float w0  = cA.z * __builtin_amdgcn_exp2f(cA.w * (dx0*dx0 + dy2));
                    float w1  = cA.z * __builtin_amdgcn_exp2f(cA.w * (dx1*dx1 + dy2));
                    a0[2*j + 0] = (_Float16)w0;
                    a1[2*j + 0] = (_Float16)w1;
                }
                {
                    float dy  = cB.y - Y;
                    float dy2 = dy * dy;
                    float dx0 = cB.x - X0;
                    float dx1 = cB.x - X1;
                    float w0  = cB.z * __builtin_amdgcn_exp2f(cB.w * (dx0*dx0 + dy2));
                    float w1  = cB.z * __builtin_amdgcn_exp2f(cB.w * (dx1*dx1 + dy2));
                    a0[2*j + 1] = (_Float16)w0;
                    a1[2*j + 1] = (_Float16)w1;
                }
            }
            acc0 = __builtin_amdgcn_wmma_f32_16x16x32_f16(
                       false, a0, false, b, (short)0, acc0, false, false);
            acc1 = __builtin_amdgcn_wmma_f32_16x16x32_f16(
                       false, a1, false, b, (short)0, acc1, false, false);
        }
    }

    // Finalize: D layout -> lane = column (channel), VGPR v = pixel row
    // (M = v for lanes 0-15, M = 8+v for lanes 16-31). den lives in column 3.
    const float addEps = 4.0f * EPS_;     // reference adds EPS once per chunk (4 chunks)
    #pragma unroll
    for (int v = 0; v < 8; ++v) {
        const int m = hi ? (8 + v) : v;
        float den0 = __shfl(acc0[v], hi ? 19 : 3, 32);
        float den1 = __shfl(acc1[v], hi ? 19 : 3, 32);
        if (col < 3) {
            int p0 = pixBase + m;
            int p1 = p0 + 16;
            float r0 = acc0[v] * __builtin_amdgcn_rcpf(fmaxf(den0 + addEps, EPS_));
            float r1 = acc1[v] * __builtin_amdgcn_rcpf(fmaxf(den1 + addEps, EPS_));
            // out[t][c][i] with t = p>>12, i = p&4095
            out[(p0 >> 12) * 12288 + col * 4096 + (p0 & 4095)] = r0;
            out[(p1 >> 12) * 12288 + col * 4096 + (p1 & 4095)] = r1;
        }
    }
}

// ---------------------------------------------------------------------------
extern "C" void kernel_launch(void* const* d_in, const int* in_sizes, int n_in,
                              void* d_out, int out_size, void* d_ws, size_t ws_size,
                              hipStream_t stream)
{
    const float* positions = (const float*)d_in[0];   // [N,3]
    const float* colors    = (const float*)d_in[1];   // [N,3]
    const float* opacities = (const float*)d_in[2];   // [N,1]
    const float* scales    = (const float*)d_in[3];   // [N,1]
    const float* qvec      = (const float*)d_in[4];   // [4]
    const float* tvec      = (const float*)d_in[5];   // [3]
    float*       out       = (float*)d_out;           // [4,3,64,64] f32

    const int N = in_sizes[2];                        // 8192
    float4* g4 = (float4*)d_ws;                       // N * 16 bytes scratch

    splat_prep_kernel<<<(N + 255) / 256, 256, 0, stream>>>(
        positions, opacities, scales, qvec, tvec, g4, N);

    // 512 waves total (16384 px / 32 px per wave); 4 waves per block
    splat_render_kernel<<<IMG_HW / (4 * 32), 128, 0, stream>>>(
        g4, colors, out, N);
}
